// SGRLayer_55551107006684
// MI455X (gfx1250) — matrix operand order repaired and checked by reading, not compiled, not run on hardware
//
#include <hip/hip_runtime.h>
#include <hip/hip_bf16.h>

// ---------------------------------------------------------------------------
// MI455X (gfx1250) implementation of the SGR layer.
// Big GEMMs run on v_wmma_f32_16x16x32_bf16 (fp32 accumulate), 32x64 tile per
// wave (8 WMMAs / k-step, loads issued ahead of compute). Small graph chain
// stays fp32 VALU. All WMMA operand loads are contiguous 16B/32B per lane,
// matching the CDNA5 A/B VGPR layouts.
// ---------------------------------------------------------------------------

#define BB 8
#define HH 96
#define WW 96
#define CC 512
#define NN (HH * WW)          // 9216
#define BN (BB * NN)          // 73728
#define MM 128
#define DD 256
#define KK 300
#define NCAT 640              // M + D + D fused output cols of GEMM1
#define SPLITS 18             // split-K for node GEMM: 9216 / 18 = 512

typedef __bf16  v16bf __attribute__((ext_vector_type(16)));
typedef __bf16  v8bf  __attribute__((ext_vector_type(8)));
typedef float   v8f   __attribute__((ext_vector_type(8)));

static __device__ inline __bf16 f2bf(float f) {
  unsigned u = __builtin_bit_cast(unsigned, f);
  unsigned r = (u + 0x7FFFu + ((u >> 16) & 1u)) >> 16;   // round-nearest-even
  unsigned short s = (unsigned short)r;
  return __builtin_bit_cast(__bf16, s);
}
static __device__ inline float bf2f(__bf16 b) {
  unsigned short s = __builtin_bit_cast(unsigned short, b);
  unsigned u = ((unsigned)s) << 16;
  return __builtin_bit_cast(float, u);
}

// A fragment (16x32 bf16, M x K): lane l holds row m=l&15,
// K in [8h,8h+8) and [16+8h,16+8h+8), h = l>>4. Two contiguous 16B loads.
static __device__ inline v16bf load_a(const __bf16* A, int lda, int m0, int k0, int lane) {
  int m = lane & 15, h = lane >> 4;
  const __bf16* p = A + (size_t)(m0 + m) * lda + k0 + 8 * h;
  v8bf lo = *(const v8bf*)(p);
  v8bf hi = *(const v8bf*)(p + 16);
  return __builtin_shufflevector(lo, hi, 0,1,2,3,4,5,6,7,8,9,10,11,12,13,14,15);
}

// B fragment (32x16 bf16, K x N) loaded from B-transposed (rows = n, contiguous K):
// lane l holds col n=l&15, K in [8h,8h+8) and [16+8h,16+8h+8). Two contiguous 16B loads.
static __device__ inline v16bf load_bt(const __bf16* Bt, int ldb, int n0, int k0, int lane) {
  int n = lane & 15, h = lane >> 4;
  const __bf16* p = Bt + (size_t)(n0 + n) * ldb + k0 + 8 * h;
  v8bf lo = *(const v8bf*)(p);
  v8bf hi = *(const v8bf*)(p + 16);
  return __builtin_shufflevector(lo, hi, 0,1,2,3,4,5,6,7,8,9,10,11,12,13,14,15);
}

static __device__ inline v8f wmma_bf16(v16bf a, v16bf b, v8f c) {
  return __builtin_amdgcn_wmma_f32_16x16x32_bf16(false, a, false, b, (short)0, c, false, false);
}

// 32x64 output tile per wave: 2 A-frags, 4 B-frags, 8 WMMAs per k-step.
// All loads for a k-step are issued before any WMMA so the scheduler can
// overlap the next step's loads with this step's matrix ops.
static __device__ inline void gemm_tile_32x64(const __bf16* __restrict__ A, int lda, int m0,
                                              const __bf16* __restrict__ Bt, int ldb, int n0,
                                              int kbeg, int kend, int lane,
                                              v8f acc[2][4]) {
  for (int k0 = kbeg; k0 < kend; k0 += 32) {
    v16bf a0 = load_a(A, lda, m0,      k0, lane);
    v16bf a1 = load_a(A, lda, m0 + 16, k0, lane);
    v16bf b0 = load_bt(Bt, ldb, n0,      k0, lane);
    v16bf b1 = load_bt(Bt, ldb, n0 + 16, k0, lane);
    v16bf b2 = load_bt(Bt, ldb, n0 + 32, k0, lane);
    v16bf b3 = load_bt(Bt, ldb, n0 + 48, k0, lane);
    acc[0][0] = wmma_bf16(a0, b0, acc[0][0]);
    acc[0][1] = wmma_bf16(a0, b1, acc[0][1]);
    acc[0][2] = wmma_bf16(a0, b2, acc[0][2]);
    acc[0][3] = wmma_bf16(a0, b3, acc[0][3]);
    acc[1][0] = wmma_bf16(a1, b0, acc[1][0]);
    acc[1][1] = wmma_bf16(a1, b1, acc[1][1]);
    acc[1][2] = wmma_bf16(a1, b2, acc[1][2]);
    acc[1][3] = wmma_bf16(a1, b3, acc[1][3]);
  }
}

// ---------------------------------------------------------------------------
// Prep kernels
// ---------------------------------------------------------------------------
__global__ __launch_bounds__(256) void k_cvt_x(const float* __restrict__ x,
                                               __bf16* __restrict__ xb) {
  size_t i = (size_t)blockIdx.x * 256 + threadIdx.x;   // BN*CC threads
  xb[i] = f2bf(x[i]);
}

// Wcat^T [640][512] bf16 = [Wv | Wt | Wql]^T
__global__ __launch_bounds__(256) void k_build_wcat(const float* __restrict__ Wv,
                                                    const float* __restrict__ Wt,
                                                    const float* __restrict__ Wql,
                                                    __bf16* __restrict__ wcat) {
  size_t i = (size_t)blockIdx.x * 256 + threadIdx.x;   // 640*512 threads
  int n = (int)(i >> 9);
  int k = (int)(i & 511);
  float v;
  if (n < MM)           v = Wv[(size_t)k * MM + n];
  else if (n < MM + DD) v = Wt[(size_t)k * DD + (n - MM)];
  else                  v = Wql[(size_t)k * DD + (n - MM - DD)];
  wcat[i] = f2bf(v);
}

// ---------------------------------------------------------------------------
// GEMM1: [BN,512] x [512,640] -> vote_logits(+bv), local(+bt), query. bf16 out.
// ---------------------------------------------------------------------------
__global__ __launch_bounds__(256) void k_gemm1(const __bf16* __restrict__ xb,
                                               const __bf16* __restrict__ wcat,
                                               const float* __restrict__ bv,
                                               const float* __restrict__ bt,
                                               __bf16* __restrict__ votel,
                                               __bf16* __restrict__ localb,
                                               __bf16* __restrict__ queryb) {
  int lane = threadIdx.x & 31, w = threadIdx.x >> 5;
  int row0 = (blockIdx.x * 8 + w) * 32;      // 2304 row-tiles of 32
  int n0g = blockIdx.y * 64;                 // 10 n-groups over 640
  v8f acc[2][4] = {};
  gemm_tile_32x64(xb, CC, row0, wcat, CC, n0g, 0, CC, lane, acc);
  int h = lane >> 4, nl = lane & 15;
#pragma unroll
  for (int t = 0; t < 4; ++t) {
    int ncol = n0g + 16 * t + nl;
    float bias; __bf16* dst; int ldd, dcol;
    if (ncol < MM)           { bias = bv[ncol];       dst = votel;  ldd = MM; dcol = ncol; }
    else if (ncol < MM + DD) { bias = bt[ncol - MM];  dst = localb; ldd = DD; dcol = ncol - MM; }
    else                     { bias = 0.0f;           dst = queryb; ldd = DD; dcol = ncol - MM - DD; }
#pragma unroll
    for (int i = 0; i < 2; ++i)
#pragma unroll
      for (int r = 0; r < 8; ++r) {
        int row = row0 + 16 * i + r + 8 * h;
        dst[(size_t)row * ldd + dcol] = f2bf(acc[i][t][r] + bias);
      }
  }
}

// ---------------------------------------------------------------------------
// Row softmax over 128 cols, bf16 in-place. One wave per row (wave32: 4/lane).
// ---------------------------------------------------------------------------
__global__ __launch_bounds__(256) void k_softmax128(__bf16* __restrict__ d) {
  int lane = threadIdx.x & 31, w = threadIdx.x >> 5;
  size_t row = (size_t)blockIdx.x * 8 + w;
  __bf16* p = d + row * MM;
  float v[4];
#pragma unroll
  for (int i = 0; i < 4; ++i) v[i] = bf2f(p[lane + 32 * i]);
  float mx = fmaxf(fmaxf(v[0], v[1]), fmaxf(v[2], v[3]));
#pragma unroll
  for (int o = 16; o >= 1; o >>= 1) mx = fmaxf(mx, __shfl_xor(mx, o, 32));
  float s = 0.f;
#pragma unroll
  for (int i = 0; i < 4; ++i) { v[i] = __expf(v[i] - mx); s += v[i]; }
#pragma unroll
  for (int o = 16; o >= 1; o >>= 1) s += __shfl_xor(s, o, 32);
  float inv = 1.0f / s;
#pragma unroll
  for (int i = 0; i < 4; ++i) p[lane + 32 * i] = f2bf(v[i] * inv);
}

// ---------------------------------------------------------------------------
// bf16 transpose (per batch z): in [R][Cc] -> out [Cc][R]
// ---------------------------------------------------------------------------
__global__ void k_transpose_bf16(const unsigned short* __restrict__ in,
                                 unsigned short* __restrict__ out, int R, int Cc) {
  __shared__ unsigned short tile[32][33];
  size_t base = (size_t)blockIdx.z * R * Cc;
  int tx = threadIdx.x, ty = threadIdx.y;
  int c = blockIdx.x * 32 + tx;
#pragma unroll
  for (int j = 0; j < 4; ++j) {
    int r = blockIdx.y * 32 + ty + 8 * j;
    tile[ty + 8 * j][tx] = in[base + (size_t)r * Cc + c];
  }
  __syncthreads();
#pragma unroll
  for (int j = 0; j < 4; ++j) {
    int orow = blockIdx.x * 32 + ty + 8 * j;
    int ocol = blockIdx.y * 32 + tx;
    out[base + (size_t)orow * R + ocol] = tile[tx][ty + 8 * j];
  }
}

// ---------------------------------------------------------------------------
// node partials: node[b] = vote[b]^T @ local[b], split-K (deterministic).
// A = voteT [M][N], Bt = localT [D][N]. partial[s][b][M][D] fp32.
// ---------------------------------------------------------------------------
__global__ __launch_bounds__(256) void k_node_partial(const __bf16* __restrict__ voteT,
                                                      const __bf16* __restrict__ localT,
                                                      float* __restrict__ partial) {
  int lane = threadIdx.x & 31, w = threadIdx.x >> 5;
  int b = blockIdx.y, s = blockIdx.z;
  int t = blockIdx.x * 8 + w;          // 0..15
  int mtile = t >> 2, dg = t & 3;      // 4 m-tiles of 32, 4 d-groups of 64
  const __bf16* A  = voteT  + (size_t)b * MM * NN;
  const __bf16* Bt = localT + (size_t)b * DD * NN;
  int kbeg = s * (NN / SPLITS);
  v8f acc[2][4] = {};
  gemm_tile_32x64(A, NN, mtile * 32, Bt, NN, dg * 64, kbeg, kbeg + NN / SPLITS, lane, acc);
  float* out = partial + ((size_t)s * BB + b) * MM * DD;
  int h = lane >> 4, nl = lane & 15;
#pragma unroll
  for (int i = 0; i < 2; ++i)
#pragma unroll
    for (int tt = 0; tt < 4; ++tt)
#pragma unroll
      for (int r = 0; r < 8; ++r)
        out[(size_t)(mtile * 32 + 16 * i + r + 8 * h) * DD + dg * 64 + 16 * tt + nl] =
            acc[i][tt][r];
}

__global__ __launch_bounds__(256) void k_node_reduce(const float* __restrict__ partial,
                                                     float* __restrict__ node) {
  size_t i = (size_t)blockIdx.x * 256 + threadIdx.x;   // B*M*D threads
  float s = 0.f;
#pragma unroll
  for (int k = 0; k < SPLITS; ++k) s += partial[(size_t)k * (BB * MM * DD) + i];
  node[i] = s;
}

// ---------------------------------------------------------------------------
// Small fp32 chain (negligible FLOPs; latency-bound, keep full precision)
// ---------------------------------------------------------------------------
__global__ __launch_bounds__(256) void k_gcn(const float* __restrict__ node,
                                             const float* __restrict__ emb,
                                             const float* __restrict__ Wg,
                                             const float* __restrict__ bg,
                                             float* __restrict__ gcn) {
  __shared__ float rowbuf[DD + KK];
  int bm = blockIdx.x;                  // b*128+m
  int m = bm & (MM - 1);
  for (int i = threadIdx.x; i < DD + KK; i += 256)
    rowbuf[i] = (i < DD) ? node[(size_t)bm * DD + i] : emb[(size_t)m * KK + (i - DD)];
  __syncthreads();
  int d = threadIdx.x;
  float s = bg[d];
  for (int k = 0; k < DD + KK; ++k) s += rowbuf[k] * Wg[(size_t)k * DD + d];
  gcn[(size_t)bm * DD + d] = s;
}

__global__ __launch_bounds__(256) void k_evolved(const float* __restrict__ adj,
                                                 const float* __restrict__ gcn,
                                                 float* __restrict__ evolved) {
  __shared__ float arow[MM];
  int bm = blockIdx.x;
  int b = bm >> 7, m = bm & (MM - 1);
  if (threadIdx.x < MM) arow[threadIdx.x] = adj[(size_t)m * MM + threadIdx.x];
  __syncthreads();
  int d = threadIdx.x;
  float s = 0.f;
  for (int k = 0; k < MM; ++k) s += arow[k] * gcn[((size_t)b * MM + k) * DD + d];
  evolved[(size_t)bm * DD + d] = fmaxf(s, 0.f);
}

__global__ __launch_bounds__(256) void k_keyv(const float* __restrict__ evolved,
                                              const float* __restrict__ Wkn,
                                              __bf16* __restrict__ keyv) {
  __shared__ float erow[DD];
  int bm = blockIdx.x;
  erow[threadIdx.x] = evolved[(size_t)bm * DD + threadIdx.x];
  __syncthreads();
  int d = threadIdx.x;
  float s = 0.f;
  for (int k = 0; k < DD; ++k) s += erow[k] * Wkn[(size_t)k * DD + d];
  keyv[(size_t)bm * DD + d] = f2bf(s);
}

// values^T stored [B][C][M] bf16 so the final GEMM's B-frag reads are contiguous
__global__ __launch_bounds__(256) void k_values(const float* __restrict__ evolved,
                                                const float* __restrict__ Wm,
                                                const float* __restrict__ bm_,
                                                __bf16* __restrict__ valuesT) {
  __shared__ float erow[DD];
  int bmi = blockIdx.x;
  int b = bmi >> 7, m = bmi & (MM - 1);
  erow[threadIdx.x] = evolved[(size_t)bmi * DD + threadIdx.x];
  __syncthreads();
  for (int c = threadIdx.x; c < CC; c += 256) {
    float s = bm_[c];
    for (int k = 0; k < DD; ++k) s += erow[k] * Wm[(size_t)k * CC + c];
    valuesT[((size_t)b * CC + c) * MM + m] = f2bf(s);
  }
}

// ---------------------------------------------------------------------------
// Attention logits: per batch [N,256] x keyv^T -> [N,128] bf16
// ---------------------------------------------------------------------------
__global__ __launch_bounds__(256) void k_attn_gemm(const __bf16* __restrict__ queryb,
                                                   const __bf16* __restrict__ keyv,
                                                   __bf16* __restrict__ attnl) {
  int lane = threadIdx.x & 31, w = threadIdx.x >> 5;
  int b = blockIdx.z;
  int row0 = (blockIdx.x * 8 + w) * 32;     // 288 row-tiles of 32 over N
  int n0 = blockIdx.y * 64;                 // 2 n-groups over 128
  const __bf16* A  = queryb + (size_t)b * NN * DD;
  const __bf16* Bt = keyv   + (size_t)b * MM * DD;
  v8f acc[2][4] = {};
  gemm_tile_32x64(A, DD, row0, Bt, DD, n0, 0, DD, lane, acc);
  __bf16* out = attnl + (size_t)b * NN * MM;
  int h = lane >> 4, nl = lane & 15;
#pragma unroll
  for (int i = 0; i < 2; ++i)
#pragma unroll
    for (int t = 0; t < 4; ++t)
#pragma unroll
      for (int r = 0; r < 8; ++r)
        out[(size_t)(row0 + 16 * i + r + 8 * h) * MM + n0 + 16 * t + nl] = f2bf(acc[i][t][r]);
}

// ---------------------------------------------------------------------------
// Final: out = relu(x + attn @ values), per batch [N,128] x [128,C]
// ---------------------------------------------------------------------------
__global__ __launch_bounds__(256) void k_final(const __bf16* __restrict__ attnb,
                                               const __bf16* __restrict__ valuesT,
                                               const float* __restrict__ x,
                                               float* __restrict__ outp) {
  int lane = threadIdx.x & 31, w = threadIdx.x >> 5;
  int b = blockIdx.z;
  int row0 = (blockIdx.x * 8 + w) * 32;     // 288 row-tiles of 32 over N
  int n0 = blockIdx.y * 64;                 // 8 n-groups over C
  const __bf16* A  = attnb   + (size_t)b * NN * MM;
  const __bf16* Bt = valuesT + (size_t)b * CC * MM;
  v8f acc[2][4] = {};
  gemm_tile_32x64(A, MM, row0, Bt, MM, n0, 0, MM, lane, acc);
  int h = lane >> 4, nl = lane & 15;
#pragma unroll
  for (int i = 0; i < 2; ++i)
#pragma unroll
    for (int t = 0; t < 4; ++t)
#pragma unroll
      for (int r = 0; r < 8; ++r) {
        size_t idx = ((size_t)b * NN + row0 + 16 * i + r + 8 * h) * CC + n0 + 16 * t + nl;
        outp[idx] = fmaxf(x[idx] + acc[i][t][r], 0.f);
      }
}

// ---------------------------------------------------------------------------
// Host launcher
// ---------------------------------------------------------------------------
extern "C" void kernel_launch(void* const* d_in, const int* in_sizes, int n_in,
                              void* d_out, int out_size, void* d_ws, size_t ws_size,
                              hipStream_t stream) {
  const float* x   = (const float*)d_in[0];
  const float* adj = (const float*)d_in[1];
  const float* emb = (const float*)d_in[2];
  const float* Wv  = (const float*)d_in[3];
  const float* bv  = (const float*)d_in[4];
  const float* Wt  = (const float*)d_in[5];
  const float* bt  = (const float*)d_in[6];
  const float* Wg  = (const float*)d_in[7];
  const float* bg  = (const float*)d_in[8];
  const float* Wql = (const float*)d_in[9];
  const float* Wkn = (const float*)d_in[10];
  const float* Wm  = (const float*)d_in[11];
  const float* bm  = (const float*)d_in[12];
  float* outp = (float*)d_out;

  char* ws = (char*)d_ws;
  size_t o = 0;
  __bf16* xb     = (__bf16*)(ws + o); o += (size_t)BN * CC * 2;            // 75.5 MB
  __bf16* wcat   = (__bf16*)(ws + o); o += (size_t)NCAT * CC * 2;          // 0.65 MB
  __bf16* votel  = (__bf16*)(ws + o); o += (size_t)BN * MM * 2;            // 18.9 MB (reused as attn logits)
  char*   region = (ws + o);          o += (size_t)BN * DD * 2;            // 37.7 MB (local, later partial/node/gcn/evolved)
  __bf16* queryb = (__bf16*)(ws + o); o += (size_t)BN * DD * 2;            // 37.7 MB
  __bf16* voteT  = (__bf16*)(ws + o); o += (size_t)BB * MM * NN * 2;       // 18.9 MB
  __bf16* localT = (__bf16*)(ws + o); o += (size_t)BB * DD * NN * 2;       // 37.7 MB
  __bf16* keyv   = (__bf16*)(ws + o); o += (size_t)BB * MM * DD * 2;       // 0.5 MB
  __bf16* valuesT= (__bf16*)(ws + o); o += (size_t)BB * CC * MM * 2;       // 1 MB
  // phase-2 reuse of "region" (local no longer needed after transpose):
  __bf16* localb  = (__bf16*)region;
  float*  partial = (float*)region;                                        // 18.9 MB
  float*  node    = (float*)(region + (size_t)SPLITS * BB * MM * DD * 4);  // 1 MB
  float*  gcn     = node + (size_t)BB * MM * DD;                           // 1 MB
  float*  evolved = gcn  + (size_t)BB * MM * DD;                           // 1 MB
  __bf16* attnl   = votel;                                                 // reuse

  // 0) precision conversion + fused weight pack
  k_cvt_x<<<(BN * CC) / 256, 256, 0, stream>>>(x, xb);
  k_build_wcat<<<(NCAT * CC) / 256, 256, 0, stream>>>(Wv, Wt, Wql, wcat);
  // 1) fused voting/local/query GEMM (48 GF, bf16 WMMA)
  k_gemm1<<<dim3(288, 10), 256, 0, stream>>>(xb, wcat, bv, bt, votel, localb, queryb);
  // 2) vote softmax + transposes feeding the node GEMM
  k_softmax128<<<BN / 8, 256, 0, stream>>>(votel);
  k_transpose_bf16<<<dim3(MM / 32, NN / 32, BB), dim3(32, 8), 0, stream>>>(
      (const unsigned short*)votel, (unsigned short*)voteT, NN, MM);
  k_transpose_bf16<<<dim3(DD / 32, NN / 32, BB), dim3(32, 8), 0, stream>>>(
      (const unsigned short*)localb, (unsigned short*)localT, NN, DD);
  // 3) node = vote^T @ local (split-K, deterministic reduction)
  k_node_partial<<<dim3(2, BB, SPLITS), 256, 0, stream>>>(voteT, localT, partial);
  k_node_reduce<<<(BB * MM * DD) / 256, 256, 0, stream>>>(partial, node);
  // 4) graph reasoning chain (fp32)
  k_gcn<<<BB * MM, 256, 0, stream>>>(node, emb, Wg, bg, gcn);
  k_evolved<<<BB * MM, 256, 0, stream>>>(adj, gcn, evolved);
  k_keyv<<<BB * MM, 256, 0, stream>>>(evolved, Wkn, keyv);
  k_values<<<BB * MM, 256, 0, stream>>>(evolved, Wm, bm, valuesT);
  // 5) cross-attention + project back
  k_attn_gemm<<<dim3(36, 2, BB), 256, 0, stream>>>(queryb, keyv, attnl);
  k_softmax128<<<BN / 8, 256, 0, stream>>>(attnl);
  k_final<<<dim3(36, 8, BB), 256, 0, stream>>>(attnl, valuesT, x, outp);

  (void)in_sizes; (void)n_in; (void)out_size; (void)ws_size;
}
